// EagerAttention_57982058496242
// MI455X (gfx1250) — matrix operand order repaired and checked by reading, not compiled, not run on hardware
//
#include <hip/hip_runtime.h>
#include <hip/hip_bf16.h>

// ---------------------------------------------------------------------------
// Eager attention block for MI455X (gfx1250, wave32, WMMA bf16 path).
//   B=2, T=2048, D=2048, H=16, KVH=4, HD=128, rep=4, theta=1e6
// Pipeline: cvt x->bf16 ; transpose weights->N-major bf16 ; 3x GEMM (WMMA,
//           async-LDS staged B tiles) ; RoPE(q,k)->bf16 head-major ; V->Vt ;
//           flash attention (WMMA) ; output GEMM (WMMA) -> fp32 d_out.
// ---------------------------------------------------------------------------

#define TSEQ   2048
#define DMODEL 2048
#define NH     16
#define NKV    4
#define HD     128
#define NB     2

typedef __bf16 bf16_t;
typedef __attribute__((ext_vector_type(8)))  __bf16 v8bf;
typedef __attribute__((ext_vector_type(16))) __bf16 v16bf;
typedef __attribute__((ext_vector_type(8)))  float  v8f;
typedef __attribute__((ext_vector_type(4)))  int    v4i;

#if __has_builtin(__builtin_amdgcn_global_load_async_to_lds_b128)
#define HAVE_ASYNC_LDS 1
#else
#define HAVE_ASYNC_LDS 0
#endif

// Build a 16-element bf16 fragment from two contiguous 8-element (16B) chunks.
__device__ __forceinline__ v16bf ld_frag(const bf16_t* p0, const bf16_t* p1) {
  v8bf lo = *reinterpret_cast<const v8bf*>(p0);
  v8bf hi = *reinterpret_cast<const v8bf*>(p1);
  v16bf r;
#pragma unroll
  for (int i = 0; i < 8; ++i) { r[i] = lo[i]; r[i + 8] = hi[i]; }
  return r;
}

// ---------------------------------------------------------------------------
// fp32 -> bf16 elementwise
// ---------------------------------------------------------------------------
__global__ __launch_bounds__(256) void k_cvt_bf16(const float* __restrict__ in,
                                                  bf16_t* __restrict__ out,
                                                  long n) {
  long i = (long)blockIdx.x * blockDim.x + threadIdx.x;
  if (i < n) out[i] = (bf16_t)in[i];
}

// ---------------------------------------------------------------------------
// W[K,N] fp32 (row-major) -> Wt[N,K] bf16 (row-major), LDS-tiled.
// ---------------------------------------------------------------------------
__global__ __launch_bounds__(256) void k_transpose(const float* __restrict__ W,
                                                   bf16_t* __restrict__ Wt,
                                                   int K, int N) {
  __shared__ float tile[32][33];
  const int nb = blockIdx.x * 32;
  const int kb = blockIdx.y * 32;
  const int tx = threadIdx.x, ty = threadIdx.y;  // blockDim = (32, 8)
#pragma unroll
  for (int i = 0; i < 32; i += 8)
    tile[ty + i][tx] = W[(size_t)(kb + ty + i) * N + nb + tx];
  __syncthreads();
#pragma unroll
  for (int i = 0; i < 32; i += 8)
    Wt[(size_t)(nb + ty + i) * K + kb + tx] = (bf16_t)tile[tx][ty + i];
}

// ---------------------------------------------------------------------------
// C[M,N] fp32 = A[M,K] bf16 @ Bt[N,K]^T bf16.
// Block = 8 waves -> 128(M) x 64(N) tile. The shared 64x64 bf16 B-chunk is
// double-buffered in LDS via async global->LDS copies (ASYNCcnt), so all
// 8 waves read B operands from LDS instead of 8x-replicated L2 traffic.
// A rows are wave-private -> direct pipelined global loads.
// ---------------------------------------------------------------------------
__global__ __launch_bounds__(256) void k_gemm_bf16(const bf16_t* __restrict__ A,
                                                   const bf16_t* __restrict__ Bt,
                                                   float* __restrict__ C,
                                                   int M, int N, int K) {
  // pitch 72 (=144B, 16B-aligned rows, 36-dword bank stride -> conflict-free)
  __shared__ __align__(16) bf16_t bbuf[2][64][72];
  const int tid  = threadIdx.x;
  const int lane = tid & 31;
  const int wave = tid >> 5;
  const int ln   = lane & 15;
  const int hi   = lane >> 4;
  const int row0 = (blockIdx.y * 8 + wave) * 16;
  const int col0 = blockIdx.x * 64;

  // Stage one 64(N) x 64(K) bf16 chunk: 512 x 16B pieces, 2 per thread.
  auto stage = [&](int buf, int k0) {
#pragma unroll
    for (int q = 0; q < 2; ++q) {
      const int p  = tid * 2 + q;
      const int n  = p >> 3;
      const int ko = (p & 7) * 8;
      const bf16_t* g = Bt + (size_t)(col0 + n) * K + k0 + ko;
#if HAVE_ASYNC_LDS
      __builtin_amdgcn_global_load_async_to_lds_b128(
          (v4i*)g, (v4i*)&bbuf[buf][n][ko], 0, 0);
#else
      *reinterpret_cast<v8bf*>(&bbuf[buf][n][ko]) =
          *reinterpret_cast<const v8bf*>(g);
#endif
    }
  };
  auto wait_stage = [&]() {
#if HAVE_ASYNC_LDS
#if __has_builtin(__builtin_amdgcn_s_wait_asynccnt)
    __builtin_amdgcn_s_wait_asynccnt(0);
#else
    asm volatile("s_wait_asynccnt 0" ::: "memory");
#endif
#endif
    __syncthreads();
  };

  v8f acc[4] = {};
  const bf16_t* arow = A + (size_t)(row0 + ln) * K;

  stage(0, 0);
  int buf = 0;
  for (int k0 = 0; k0 < K; k0 += 64) {
    wait_stage();                     // current buffer resident in LDS (all waves)
    if (k0 + 64 < K) stage(buf ^ 1, k0 + 64);  // prefetch next chunk async
    __builtin_prefetch(arow + k0 + 128, 0, 1); // global_prefetch_b8 for A stream
#pragma unroll
    for (int kk = 0; kk < 64; kk += 32) {
      const bf16_t* ap = arow + k0 + kk + hi * 8;
      v16bf af = ld_frag(ap, ap + 16);
#pragma unroll
      for (int j = 0; j < 4; ++j) {
        const bf16_t* bp = &bbuf[buf][j * 16 + ln][kk + hi * 8];
        v16bf bfg = ld_frag(bp, bp + 16);
        acc[j] = __builtin_amdgcn_wmma_f32_16x16x32_bf16(
            false, af, false, bfg, (short)0, acc[j], false, false);
      }
    }
    buf ^= 1;
  }
#pragma unroll
  for (int j = 0; j < 4; ++j)
#pragma unroll
    for (int r = 0; r < 8; ++r)
      C[(size_t)(row0 + hi * 8 + r) * N + col0 + j * 16 + ln] = acc[j][r];
}

// ---------------------------------------------------------------------------
// RoPE: in[B,T,heads,128] fp32 -> out[B,heads,T,128] bf16
// ---------------------------------------------------------------------------
__global__ __launch_bounds__(256) void k_rope(const float* __restrict__ in,
                                              bf16_t* __restrict__ out,
                                              int heads, long total) {
  long idx = (long)blockIdx.x * blockDim.x + threadIdx.x;
  if (idx >= total) return;
  const int d = (int)(idx & 127);
  long r = idx >> 7;
  const int h = (int)(r % heads); r /= heads;
  const int t = (int)(r % TSEQ);
  const int b = (int)(r / TSEQ);

  const int i = d & 63;  // inv_freq index
  // inv_freq[i] = theta^(-2i/128), theta=1e6, ln(1e6)=13.815510557964274
  const float inv = __expf(-(float)(2 * i) * (13.815510557964274f / 128.0f));
  const float ang = (float)t * inv;
  float s, c;
  __sincosf(ang, &s, &c);
  const float v0 = in[idx];
  const float partner = (d < 64) ? -in[idx + 64] : in[idx - 64];
  out[(((size_t)b * heads + h) * TSEQ + t) * HD + d] = (bf16_t)(v0 * c + partner * s);
}

// ---------------------------------------------------------------------------
// V transpose: vf[B,T,KVH,128] fp32 -> vT[B,KVH,128,T] bf16
// ---------------------------------------------------------------------------
__global__ __launch_bounds__(256) void k_vt(const float* __restrict__ vf,
                                            bf16_t* __restrict__ vT, long total) {
  long idx = (long)blockIdx.x * blockDim.x + threadIdx.x;
  if (idx >= total) return;
  const int d = (int)(idx & 127);
  long r = idx >> 7;
  const int kv = (int)(r & 3); r >>= 2;
  const int t  = (int)(r % TSEQ);
  const int b  = (int)(r / TSEQ);
  vT[(((size_t)b * NKV + kv) * HD + d) * TSEQ + t] = (bf16_t)vf[idx];
}

// ---------------------------------------------------------------------------
// Flash attention: one wave per 16-query tile; 32-key blocks; online softmax.
// qb[B,H,T,128] bf16 (rope'd), kb[B,KVH,T,128] bf16 (rope'd), vT[B,KVH,128,T].
// Output: attnb[B,T,H*128] bf16 (token-major for the final GEMM).
// ---------------------------------------------------------------------------
__global__ __launch_bounds__(256) void k_attn(const bf16_t* __restrict__ qb,
                                              const bf16_t* __restrict__ kb,
                                              const bf16_t* __restrict__ vT,
                                              bf16_t* __restrict__ attnb) {
  __shared__ __align__(16) bf16_t pstage[8][16][48];  // 48-pitch: 16B-aligned
  const int lane = threadIdx.x & 31;
  const int wave = threadIdx.x >> 5;
  const int ln   = lane & 15;
  const int hi   = lane >> 4;
  const int h  = blockIdx.y;
  const int b  = blockIdx.z;
  const int kv = h >> 2;  // rep = NH/NKV = 4
  const int q0 = (blockIdx.x * 8 + wave) * 16;
  const float scale = 0.08838834764831845f;  // 1/sqrt(128)

  const bf16_t* Q  = qb + ((size_t)(b * NH + h) * TSEQ + q0) * HD;
  const bf16_t* Kt = kb + (size_t)(b * NKV + kv) * TSEQ * HD;
  const bf16_t* Vt = vT + (size_t)(b * NKV + kv) * HD * TSEQ;

  // Q A-fragments for the whole tile (16 x 128 = 4 chunks of K=32)
  v16bf qfrag[4];
#pragma unroll
  for (int c = 0; c < 4; ++c) {
    const bf16_t* ap = Q + ln * HD + c * 32 + hi * 8;
    qfrag[c] = ld_frag(ap, ap + 16);
  }

  v8f oacc[8] = {};
  float mrow[8], lrow[8];
#pragma unroll
  for (int r = 0; r < 8; ++r) { mrow[r] = -3.0e38f; lrow[r] = 0.f; }

  const int kend = q0 + 16;  // causal: keys <= q0+15
  for (int s0 = 0; s0 < kend; s0 += 32) {
    // S tiles: keys [s0, s0+16) and [s0+16, s0+32)
    v8f sacc[2] = {};
#pragma unroll
    for (int c = 0; c < 4; ++c) {
#pragma unroll
      for (int t = 0; t < 2; ++t) {
        const bf16_t* bp = Kt + (size_t)(s0 + t * 16 + ln) * HD + c * 32 + hi * 8;
        v16bf kf = ld_frag(bp, bp + 16);
        sacc[t] = __builtin_amdgcn_wmma_f32_16x16x32_bf16(
            false, qfrag[c], false, kf, (short)0, sacc[t], false, false);
      }
    }

    // Online softmax. C-layout: reg r <-> row (hi*8 + r), lane&15 <-> key col.
    float alpha[8];
#pragma unroll
    for (int r = 0; r < 8; ++r) {
      const int m  = hi * 8 + r;
      const int qi = q0 + m;
      float s0v = sacc[0][r] * scale;
      float s1v = sacc[1][r] * scale;
      if (s0 + ln      > qi) s0v = -3.0e38f;
      if (s0 + 16 + ln > qi) s1v = -3.0e38f;
      float mx = fmaxf(s0v, s1v);
#pragma unroll
      for (int off = 1; off < 16; off <<= 1)
        mx = fmaxf(mx, __shfl_xor(mx, off, 16));
      const float mnew = fmaxf(mrow[r], mx);
      const float a  = __expf(mrow[r] - mnew);
      const float p0 = __expf(s0v - mnew);
      const float p1 = __expf(s1v - mnew);
      float rs = p0 + p1;
#pragma unroll
      for (int off = 1; off < 16; off <<= 1)
        rs += __shfl_xor(rs, off, 16);
      lrow[r] = lrow[r] * a + rs;
      mrow[r] = mnew;
      alpha[r] = a;
      // Stage P (C-layout -> LDS) so it can be re-read in A-fragment layout.
      pstage[wave][m][ln]      = (bf16_t)p0;
      pstage[wave][m][ln + 16] = (bf16_t)p1;
    }
    asm volatile("s_wait_dscnt 0" ::: "memory");  // same-wave LDS RAW
    v16bf pfrag = ld_frag(&pstage[wave][ln][hi * 8],
                          &pstage[wave][ln][16 + hi * 8]);

    // Rescale running output by alpha, then accumulate P @ V.
#pragma unroll
    for (int db = 0; db < 8; ++db)
#pragma unroll
      for (int r = 0; r < 8; ++r) oacc[db][r] *= alpha[r];
#pragma unroll
    for (int db = 0; db < 8; ++db) {
      const bf16_t* vp = Vt + (size_t)(db * 16 + ln) * TSEQ + s0 + hi * 8;
      v16bf vfr = ld_frag(vp, vp + 16);
      oacc[db] = __builtin_amdgcn_wmma_f32_16x16x32_bf16(
          false, pfrag, false, vfr, (short)0, oacc[db], false, false);
    }
  }

  // Epilogue: normalize by l, store bf16 token-major [B,T, H*128].
#pragma unroll
  for (int r = 0; r < 8; ++r) {
    const float inv_l = 1.0f / lrow[r];
    const int row = q0 + hi * 8 + r;
    bf16_t* orow = attnb + ((size_t)b * TSEQ + row) * DMODEL + h * HD;
#pragma unroll
    for (int db = 0; db < 8; ++db)
      orow[db * 16 + ln] = (bf16_t)(oacc[db][r] * inv_l);
  }
}

// ---------------------------------------------------------------------------
// Host-side orchestration
// ---------------------------------------------------------------------------
extern "C" void kernel_launch(void* const* d_in, const int* in_sizes, int n_in,
                              void* d_out, int out_size, void* d_ws, size_t ws_size,
                              hipStream_t stream) {
  (void)in_sizes; (void)n_in; (void)out_size; (void)ws_size;
  const float* x  = (const float*)d_in[0];
  const float* wq = (const float*)d_in[1];
  const float* wk = (const float*)d_in[2];
  const float* wv = (const float*)d_in[3];
  const float* wo = (const float*)d_in[4];
  float* y = (float*)d_out;

  char* base = (char*)d_ws;
  size_t off = 0;
  auto carve = [&](size_t bytes) -> void* {
    void* p = base + off;
    off = (off + bytes + 255) & ~(size_t)255;
    return p;
  };
  const size_t BT = (size_t)NB * TSEQ;  // 4096 token rows
  bf16_t* xb   = (bf16_t*)carve(BT * DMODEL * 2);
  bf16_t* wqT  = (bf16_t*)carve((size_t)DMODEL * DMODEL * 2);
  bf16_t* wkT  = (bf16_t*)carve((size_t)512 * DMODEL * 2);
  bf16_t* wvT  = (bf16_t*)carve((size_t)512 * DMODEL * 2);
  bf16_t* woT  = (bf16_t*)carve((size_t)DMODEL * DMODEL * 2);
  float*  qf   = (float*)carve(BT * DMODEL * 4);
  float*  kf   = (float*)carve(BT * 512 * 4);
  float*  vf   = (float*)carve(BT * 512 * 4);
  bf16_t* qb2  = (bf16_t*)carve(BT * DMODEL * 2);
  bf16_t* kb2  = (bf16_t*)carve(BT * 512 * 2);
  bf16_t* vTb  = (bf16_t*)carve(BT * 512 * 2);
  bf16_t* attn = (bf16_t*)carve(BT * DMODEL * 2);

  // 1) x -> bf16
  k_cvt_bf16<<<(int)((BT * DMODEL + 255) / 256), 256, 0, stream>>>(x, xb, (long)(BT * DMODEL));

  // 2) weights -> N-major bf16
  dim3 tb(32, 8);
  k_transpose<<<dim3(DMODEL / 32, DMODEL / 32), tb, 0, stream>>>(wq, wqT, DMODEL, DMODEL);
  k_transpose<<<dim3(512 / 32,    DMODEL / 32), tb, 0, stream>>>(wk, wkT, DMODEL, 512);
  k_transpose<<<dim3(512 / 32,    DMODEL / 32), tb, 0, stream>>>(wv, wvT, DMODEL, 512);
  k_transpose<<<dim3(DMODEL / 32, DMODEL / 32), tb, 0, stream>>>(wo, woT, DMODEL, DMODEL);

  // 3) QKV projections (WMMA, async-LDS B staging)
  k_gemm_bf16<<<dim3(DMODEL / 64, BT / 128), 256, 0, stream>>>(xb, wqT, qf, (int)BT, DMODEL, DMODEL);
  k_gemm_bf16<<<dim3(512 / 64,    BT / 128), 256, 0, stream>>>(xb, wkT, kf, (int)BT, 512, DMODEL);
  k_gemm_bf16<<<dim3(512 / 64,    BT / 128), 256, 0, stream>>>(xb, wvT, vf, (int)BT, 512, DMODEL);

  // 4) RoPE + layout to head-major bf16; V transpose
  k_rope<<<(int)((BT * DMODEL + 255) / 256), 256, 0, stream>>>(qf, qb2, NH,  (long)(BT * DMODEL));
  k_rope<<<(int)((BT * 512 + 255) / 256),    256, 0, stream>>>(kf, kb2, NKV, (long)(BT * 512));
  k_vt  <<<(int)((BT * 512 + 255) / 256),    256, 0, stream>>>(vf, vTb, (long)(BT * 512));

  // 5) Flash attention (WMMA)
  k_attn<<<dim3(TSEQ / 128, NH, NB), 256, 0, stream>>>(qb2, kb2, vTb, attn);

  // 6) Output projection (WMMA) -> fp32 result
  k_gemm_bf16<<<dim3(DMODEL / 64, BT / 128), 256, 0, stream>>>(attn, woT, y, (int)BT, DMODEL, DMODEL);
}
